// ANFSAT_39187281609014
// MI455X (gfx1250) — matrix operand order repaired and checked by reading, not compiled, not run on hardware
//
#include <hip/hip_runtime.h>
#include <stdint.h>

// ---------------------------------------------------------------------------
// Problem constants (match the reference)
// ---------------------------------------------------------------------------
#define FEAT    128
#define NV      30000
#define NPAD    30016          // NV padded to multiple of 64 (GEMM tile M)
#define NC      60000
#define NA      300000
#define ROUNDS  8
#define PI_F    3.14159265358979f

typedef __attribute__((ext_vector_type(16))) _Float16 v16h;
typedef __attribute__((ext_vector_type(8)))  float    v8f;

// ---------------------------------------------------------------------------
// WMMA GEMM:  C[M,Nout] = act( A[M,K](f16) * Wt[Nout,K]^T (f16) + bias )
//   - A row-major [M,K] f16, K multiple of 32, M multiple of 64 (padded)
//   - Wt row-major [Nout,K] f16 (weights pre-transposed), Nout multiple of 64
//   - block: 64x64 tile, 128 threads = 4 wave32, each wave 32x32 (2x2 WMMA)
//   - branch-free hot path; staging loads batched for memory-level parallelism
// ---------------------------------------------------------------------------
#define BM 64
#define BN 64
#define BK 32
#define SA 40   // LDS row stride in halfs; 80B rows keep 16B chunks aligned
#define SB 40

union Frag16 { v16h h; uint32_t u[8]; };

__global__ __launch_bounds__(128)
void gemm_wmma_f16(const _Float16* __restrict__ A,
                   const _Float16* __restrict__ Wt,
                   const float*    __restrict__ bias,
                   _Float16*       __restrict__ C16,
                   float*          __restrict__ C32,
                   int M, int K, int Nout, int relu)
{
    __shared__ __attribute__((aligned(16))) _Float16 lA[BM * SA];
    __shared__ __attribute__((aligned(16))) _Float16 lB[BN * SB];

    const int tid  = threadIdx.x;
    const int lane = tid & 31;
    const int wave = tid >> 5;
    const int wm   = (wave >> 1) * 32;   // wave row offset in block tile
    const int wn   = (wave & 1)  * 32;   // wave col offset in block tile
    const int m0   = blockIdx.x * BM;
    const int n0   = blockIdx.y * BN;

    const int half = lane >> 4;          // 0: lanes 0-15, 1: lanes 16-31
    const int l15  = lane & 15;

    v8f acc[2][2];
    const v8f vzero = {0.f,0.f,0.f,0.f,0.f,0.f,0.f,0.f};
    acc[0][0] = vzero; acc[0][1] = vzero; acc[1][0] = vzero; acc[1][1] = vzero;

    // staging geometry: each tile = 64 rows x 32 halfs = 256 x 16B chunks,
    // 128 threads stage a tile in 2 x b128 per thread (4 chunks per row)
    const int sr  = tid >> 2;            // row for chunk 0 (0..31)
    const int skk = (tid & 3) << 3;      // half offset for chunk

    for (int k0 = 0; k0 < K; k0 += BK) {
        // --- issue all 4 staging loads first (batched, overlapping in flight)
        const uint4 va0 = *reinterpret_cast<const uint4*>(
            A  + (size_t)(m0 + sr)      * K + k0 + skk);
        const uint4 va1 = *reinterpret_cast<const uint4*>(
            A  + (size_t)(m0 + sr + 32) * K + k0 + skk);
        const uint4 vb0 = *reinterpret_cast<const uint4*>(
            Wt + (size_t)(n0 + sr)      * K + k0 + skk);
        const uint4 vb1 = *reinterpret_cast<const uint4*>(
            Wt + (size_t)(n0 + sr + 32) * K + k0 + skk);

        __syncthreads();   // previous iteration's fragment reads are done
        *reinterpret_cast<uint4*>(&lA[(sr)      * SA + skk]) = va0;
        *reinterpret_cast<uint4*>(&lA[(sr + 32) * SA + skk]) = va1;
        *reinterpret_cast<uint4*>(&lB[(sr)      * SB + skk]) = vb0;
        *reinterpret_cast<uint4*>(&lB[(sr + 32) * SB + skk]) = vb1;

        if (k0 + BK < K) {   // prefetch next K tile (global_prefetch_b8)
            __builtin_prefetch(A  + (size_t)(m0 + (tid >> 1)) * K + k0 + BK, 0, 1);
            __builtin_prefetch(Wt + (size_t)(n0 + (tid >> 1)) * K + k0 + BK, 0, 1);
        }
        __syncthreads();

        const uint32_t* lAu = reinterpret_cast<const uint32_t*>(lA);
        const uint32_t* lBu = reinterpret_cast<const uint32_t*>(lB);

        // --- build fragments per ISA 16-bit A(16x32)/B(32x16) layouts
        Frag16 a[2], b[2];
#pragma unroll
        for (int t = 0; t < 2; ++t) {
            const int m = wm + t * 16 + l15;
#pragma unroll
            for (int p = 0; p < 8; ++p) {
                // A: VGPR p holds K = {2p | 2p+8 (p>=4)} + 8*half, and K+1
                int kb = ((p < 4) ? (2 * p) : (2 * p + 8)) + 8 * half;
                a[t].u[p] = lAu[(m * SA + kb) >> 1];
            }
            const int n = wn + t * 16 + l15;
#pragma unroll
            for (int v = 0; v < 8; ++v) {
                // B: VGPR v holds K = {2v | 2v+8 (v>=4)} + 8*half, and K+1 (column = l15)
                int kb = ((v < 4) ? (2 * v) : (2 * v + 8)) + 8 * half;
                b[t].u[v] = lBu[(n * SB + kb) >> 1];
            }
        }
#pragma unroll
        for (int tm = 0; tm < 2; ++tm)
#pragma unroll
            for (int tn = 0; tn < 2; ++tn)
                acc[tm][tn] = __builtin_amdgcn_wmma_f32_16x16x32_f16(
                    false, a[tm].h, false, b[tn].h,
                    (short)0, acc[tm][tn], false, false);
    }

    // --- epilogue: C/D layout -> lane(0-15): N=lane, VGPR r: M=r; lanes16-31: M=8+r
#pragma unroll
    for (int tm = 0; tm < 2; ++tm) {
        const int rbase = m0 + wm + tm * 16 + (half ? 8 : 0);
#pragma unroll
        for (int tn = 0; tn < 2; ++tn) {
            const int col = n0 + wn + tn * 16 + l15;
            const float bv = bias ? bias[col] : 0.f;
#pragma unroll
            for (int r = 0; r < 8; ++r) {
                const int row = rbase + r;
                float val = acc[tm][tn][r] + bv;
                if (relu) val = fmaxf(val, 0.f);
                if (C16) C16[(size_t)row * Nout + col] = (_Float16)val;
                if (C32) C32[(size_t)row * Nout + col] = val;
            }
        }
    }
}

// ---------------------------------------------------------------------------
// Support kernels
// ---------------------------------------------------------------------------
__global__ void zero_f32(float* p, long n) {
    long i = (long)blockIdx.x * blockDim.x + threadIdx.x;
    if (i < n) p[i] = 0.f;
}
__global__ void zero_f16(_Float16* p, long n) {
    long i = (long)blockIdx.x * blockDim.x + threadIdx.x;
    if (i < n) p[i] = (_Float16)0.f;
}

// W[K0,Nout] f32 -> Wt[Nout,Kp] f16 (transposed, zero-padded K)
__global__ void prep_weight(const float* __restrict__ W, _Float16* __restrict__ Wt,
                            int K0, int Kp, int Nout) {
    long i = (long)blockIdx.x * blockDim.x + threadIdx.x;
    if (i >= (long)Nout * Kp) return;
    int n = (int)(i / Kp), k = (int)(i % Kp);
    Wt[i] = (k < K0) ? (_Float16)W[(size_t)k * Nout + n] : (_Float16)0.f;
}

// zero_state: onehot(0) - 1/f, scaled by sqrt(f)*0.25  (same every row)
__global__ void init_vars(float* __restrict__ vars, int M) {
    long i = (long)blockIdx.x * blockDim.x + threadIdx.x;
    if (i >= (long)M * FEAT) return;
    int c = (int)(i & (FEAT - 1));
    const float s = 2.82842712474619f; // sqrt(128)*0.25
    vars[i] = ((c == 0) ? (1.f - 1.f / FEAT) : (-1.f / FEAT)) * s;
}

// qin[M,160] f16 = [vars(128) | noise_r(4) | 0(28)]
__global__ void build_qin(const float* __restrict__ vars,
                          const float* __restrict__ noise_r,
                          _Float16* __restrict__ qin, int M) {
    long i = (long)blockIdx.x * blockDim.x + threadIdx.x;
    if (i >= (long)M * 160) return;
    int row = (int)(i / 160), c = (int)(i % 160);
    float v = 0.f;
    if (c < 128)      v = vars[(size_t)row * FEAT + c];
    else if (c < 132) v = noise_r[(size_t)row * 4 + (c - 128)];
    qin[i] = (_Float16)v;
}

__global__ void sigmoid_k(const float* __restrict__ q, float* __restrict__ p, long n) {
    long i = (long)blockIdx.x * blockDim.x + threadIdx.x;
    if (i < n) p[i] = 1.f / (1.f + __expf(-q[i]));
}

// s[ci] += PI * p[i1]*p[i2]   (2 ands per 256-thread block, 128 channels each)
__global__ void and_fwd(const int* __restrict__ i1, const int* __restrict__ i2,
                        const int* __restrict__ ci, const float* __restrict__ p,
                        float* __restrict__ s, int n_ands) {
    int t = blockIdx.x * 2 + (threadIdx.x >> 7);
    int c = threadIdx.x & 127;
    if (t >= n_ands) return;
    int a = i1[t], b = i2[t], cl = ci[t];
    float v = p[(size_t)a * FEAT + c] * p[(size_t)b * FEAT + c];
    atomicAdd(&s[(size_t)cl * FEAT + c], PI_F * v);
}

// s := -PI * sin(s)   (dL/da for each AND's clause)
__global__ void clause_grad(float* __restrict__ s, long n) {
    long i = (long)blockIdx.x * blockDim.x + threadIdx.x;
    if (i < n) s[i] = -PI_F * __sinf(s[i]);
}

// gradp[i1] += g*p[i2]; gradp[i2] += g*p[i1]
__global__ void and_bwd(const int* __restrict__ i1, const int* __restrict__ i2,
                        const int* __restrict__ ci, const float* __restrict__ p,
                        const float* __restrict__ s, float* __restrict__ gp, int n_ands) {
    int t = blockIdx.x * 2 + (threadIdx.x >> 7);
    int c = threadIdx.x & 127;
    if (t >= n_ands) return;
    int a = i1[t], b = i2[t], cl = ci[t];
    float g  = s[(size_t)cl * FEAT + c];
    float pa = p[(size_t)a * FEAT + c];
    float pb = p[(size_t)b * FEAT + c];
    atomicAdd(&gp[(size_t)a * FEAT + c], g * pb);
    atomicAdd(&gp[(size_t)b * FEAT + c], g * pa);
}

// unit[M,256] f16 = [ gradp*p*(1-p) (128) | vars (128) ]
__global__ void build_unit(const float* __restrict__ gp, const float* __restrict__ p,
                           const float* __restrict__ vars, _Float16* __restrict__ unit,
                           int M) {
    long i = (long)blockIdx.x * blockDim.x + threadIdx.x;
    if (i >= (long)M * FEAT) return;
    int row = (int)(i >> 7), c = (int)(i & 127);
    float pv = p[i];
    float gq = gp[i] * pv * (1.f - pv);
    unit[(size_t)row * 256 + c]       = (_Float16)gq;
    unit[(size_t)row * 256 + 128 + c] = (_Float16)vars[i];
}

// PairNorm stats: per-column sums over 128-row slabs
__global__ void colsum_k(const float* __restrict__ x, float* __restrict__ colsum, int M) {
    int c = threadIdx.x;                  // 128 threads
    int r0 = blockIdx.x * 128;
    float s = 0.f;
    for (int r = 0; r < 128; ++r) {
        int row = r0 + r;
        if (row < M) s += x[(size_t)row * FEAT + c];
    }
    atomicAdd(&colsum[c], s);
}

__global__ void sqsum_k(const float* __restrict__ x, const float* __restrict__ colsum,
                        float* __restrict__ sq, int M) {
    __shared__ float red[128];
    int c = threadIdx.x;
    float mean = colsum[c] * (1.f / (float)M);
    int r0 = blockIdx.x * 128;
    float s = 0.f;
    for (int r = 0; r < 128; ++r) {
        int row = r0 + r;
        if (row < M) { float d = x[(size_t)row * FEAT + c] - mean; s += d * d; }
    }
    red[c] = s; __syncthreads();
    for (int o = 64; o > 0; o >>= 1) {
        if (c < o) red[c] += red[c + o];
        __syncthreads();
    }
    if (c == 0) atomicAdd(sq, red[0]);
}

// vars = (x-mean)/sqrt(sq/M+eps)*0.25 + 0.1*vars ; also f16 copy
__global__ void pairnorm_fin(const float* __restrict__ x, const float* __restrict__ colsum,
                             const float* __restrict__ sq, float* __restrict__ vars,
                             _Float16* __restrict__ v16, int M) {
    long i = (long)blockIdx.x * blockDim.x + threadIdx.x;
    if (i >= (long)M * FEAT) return;
    int c = (int)(i & 127);
    float mean  = colsum[c] * (1.f / (float)M);
    float scale = sqrtf(sq[0] * (1.f / (float)M) + 1e-6f);
    float nv = (x[i] - mean) / scale * 0.25f;
    float v  = nv + 0.1f * vars[i];
    vars[i] = v;
    v16[i]  = (_Float16)v;
}

// final 128->1 dot + bias, stable BCE vs smoothed labels; accumulate loss
__global__ void logit_loss_k(const _Float16* __restrict__ h, const float* __restrict__ w,
                             const float* __restrict__ b, const int* __restrict__ labels,
                             float* __restrict__ out, int M, int write_logits) {
    __shared__ float wsh[128];
    __shared__ float red[128];
    int tid = threadIdx.x;
    wsh[tid] = w[tid];
    __syncthreads();
    int row = blockIdx.x * 128 + tid;
    float l = 0.f;
    if (row < M) {
        float acc = b[0];
        for (int c = 0; c < 128; ++c)
            acc += (float)h[(size_t)row * FEAT + c] * wsh[c];
        float sm = 0.05f + 0.9f * (float)labels[row];
        l = fmaxf(acc, 0.f) - acc * sm + log1pf(__expf(-fabsf(acc)));
        if (write_logits) out[row] = acc;
    }
    red[tid] = l; __syncthreads();
    for (int o = 64; o > 0; o >>= 1) {
        if (tid < o) red[tid] += red[tid + o];
        __syncthreads();
    }
    if (tid == 0) atomicAdd(&out[M], red[0] * (1.f / ((float)M * (float)ROUNDS)));
}

// ---------------------------------------------------------------------------
// Host driver
// ---------------------------------------------------------------------------
static inline long cdivl(long a, long b) { return (a + b - 1) / b; }

extern "C" void kernel_launch(void* const* d_in, const int* in_sizes, int n_in,
                              void* d_out, int out_size, void* d_ws, size_t ws_size,
                              hipStream_t stream) {
    (void)in_sizes; (void)n_in; (void)out_size; (void)ws_size;

    const int*   i1     = (const int*)  d_in[0];
    const int*   i2     = (const int*)  d_in[1];
    const int*   ci     = (const int*)  d_in[2];
    const int*   labels = (const int*)  d_in[3];
    const float* noise  = (const float*)d_in[4];
    // d_in[5]=n_vars, d_in[6]=n_clauses (compile-time constants here)
    const float* qW0 = (const float*)d_in[7];  const float* qb0 = (const float*)d_in[8];
    const float* qW1 = (const float*)d_in[9];  const float* qb1 = (const float*)d_in[10];
    const float* qW2 = (const float*)d_in[11]; const float* qb2 = (const float*)d_in[12];
    const float* gW0 = (const float*)d_in[13]; const float* gb0 = (const float*)d_in[14];
    const float* gW1 = (const float*)d_in[15]; const float* gb1 = (const float*)d_in[16];
    const float* gW2 = (const float*)d_in[17]; const float* gb2 = (const float*)d_in[18];
    const float* oW0 = (const float*)d_in[19]; const float* ob0 = (const float*)d_in[20];
    const float* oW1 = (const float*)d_in[21]; const float* ob1 = (const float*)d_in[22];
    const float* oW2 = (const float*)d_in[23]; const float* ob2 = (const float*)d_in[24];

    float* out = (float*)d_out;

    // --- workspace bump allocator (activation buffers padded to NPAD rows)
    char* ws = (char*)d_ws;
    size_t off = 0;
    auto alloc = [&](size_t bytes) -> char* {
        char* p = ws + off;
        off = (off + bytes + 255) & ~(size_t)255;
        return p;
    };
    _Float16* bufIn  = (_Float16*)alloc((size_t)NPAD * 256 * 2); // qin(160)/unit(256)/var16(128)
    _Float16* bufA   = (_Float16*)alloc((size_t)NPAD * 384 * 2);
    _Float16* bufB   = (_Float16*)alloc((size_t)NPAD * 384 * 2);
    float*    query  = (float*)   alloc((size_t)NPAD * FEAT * 4);
    float*    pbuf   = (float*)   alloc((size_t)NV   * FEAT * 4);
    float*    sbuf   = (float*)   alloc((size_t)NC   * FEAT * 4);
    float*    gradp  = (float*)   alloc((size_t)NV   * FEAT * 4);
    float*    gv     = (float*)   alloc((size_t)NPAD * FEAT * 4);
    float*    vars   = (float*)   alloc((size_t)NV   * FEAT * 4);
    float*    cstats = (float*)   alloc(129 * 4);   // colsum[128], sqsum at [128]
    _Float16* qWt0 = (_Float16*)alloc((size_t)384 * 160 * 2);
    _Float16* qWt1 = (_Float16*)alloc((size_t)384 * 384 * 2);
    _Float16* qWt2 = (_Float16*)alloc((size_t)128 * 384 * 2);
    _Float16* gWt0 = (_Float16*)alloc((size_t)256 * 256 * 2);
    _Float16* gWt1 = (_Float16*)alloc((size_t)256 * 256 * 2);
    _Float16* gWt2 = (_Float16*)alloc((size_t)128 * 256 * 2);
    _Float16* oWt0 = (_Float16*)alloc((size_t)128 * 128 * 2);
    _Float16* oWt1 = (_Float16*)alloc((size_t)128 * 128 * 2);

    const int TB = 256;
    auto g1d = [&](long n) { return dim3((unsigned)cdivl(n, TB)); };

    // --- one-time per launch: weights -> f16 transposed+padded, init state,
    //     zero f16 activation buffers so padded rows stay finite forever
    zero_f16<<<g1d((long)NPAD*256), TB, 0, stream>>>(bufIn, (long)NPAD*256);
    zero_f16<<<g1d((long)NPAD*384), TB, 0, stream>>>(bufA,  (long)NPAD*384);
    zero_f16<<<g1d((long)NPAD*384), TB, 0, stream>>>(bufB,  (long)NPAD*384);
    prep_weight<<<g1d(384L*160), TB, 0, stream>>>(qW0, qWt0, 132, 160, 384);
    prep_weight<<<g1d(384L*384), TB, 0, stream>>>(qW1, qWt1, 384, 384, 384);
    prep_weight<<<g1d(128L*384), TB, 0, stream>>>(qW2, qWt2, 384, 384, 128);
    prep_weight<<<g1d(256L*256), TB, 0, stream>>>(gW0, gWt0, 256, 256, 256);
    prep_weight<<<g1d(256L*256), TB, 0, stream>>>(gW1, gWt1, 256, 256, 256);
    prep_weight<<<g1d(128L*256), TB, 0, stream>>>(gW2, gWt2, 256, 256, 128);
    prep_weight<<<g1d(128L*128), TB, 0, stream>>>(oW0, oWt0, 128, 128, 128);
    prep_weight<<<g1d(128L*128), TB, 0, stream>>>(oW1, oWt1, 128, 128, 128);
    init_vars<<<g1d((long)NV*FEAT), TB, 0, stream>>>(vars, NV);
    zero_f32<<<1, 1, 0, stream>>>(out + NV, 1);   // loss accumulator

    const dim3 gemmB(128);
    const unsigned GX = NPAD / BM;   // 469, exact

    for (int r = 0; r < ROUNDS; ++r) {
        const float* noise_r = noise + (size_t)r * NV * 4;

        // ---- query MLP: [vars|z] (160) -> 384 -> 384 -> 128
        build_qin<<<g1d((long)NV*160), TB, 0, stream>>>(vars, noise_r, bufIn, NV);
        gemm_wmma_f16<<<dim3(GX, 6), gemmB, 0, stream>>>(bufIn, qWt0, qb0, bufA, nullptr, NPAD, 160, 384, 1);
        gemm_wmma_f16<<<dim3(GX, 6), gemmB, 0, stream>>>(bufA,  qWt1, qb1, bufB, nullptr, NPAD, 384, 384, 1);
        gemm_wmma_f16<<<dim3(GX, 2), gemmB, 0, stream>>>(bufB,  qWt2, qb2, nullptr, query, NPAD, 384, 128, 0);

        // ---- graph grad step
        sigmoid_k<<<g1d((long)NV*FEAT), TB, 0, stream>>>(query, pbuf, (long)NV*FEAT);
        zero_f32<<<g1d((long)NC*FEAT), TB, 0, stream>>>(sbuf, (long)NC*FEAT);
        and_fwd<<<dim3(NA/2), dim3(256), 0, stream>>>(i1, i2, ci, pbuf, sbuf, NA);
        clause_grad<<<g1d((long)NC*FEAT), TB, 0, stream>>>(sbuf, (long)NC*FEAT);
        zero_f32<<<g1d((long)NV*FEAT), TB, 0, stream>>>(gradp, (long)NV*FEAT);
        and_bwd<<<dim3(NA/2), dim3(256), 0, stream>>>(i1, i2, ci, pbuf, sbuf, gradp, NA);
        build_unit<<<g1d((long)NV*FEAT), TB, 0, stream>>>(gradp, pbuf, vars, bufIn, NV);

        // ---- gate MLP: 256 -> 256 -> 256 -> 128
        gemm_wmma_f16<<<dim3(GX, 4), gemmB, 0, stream>>>(bufIn, gWt0, gb0, bufA, nullptr, NPAD, 256, 256, 1);
        gemm_wmma_f16<<<dim3(GX, 4), gemmB, 0, stream>>>(bufA,  gWt1, gb1, bufB, nullptr, NPAD, 256, 256, 1);
        gemm_wmma_f16<<<dim3(GX, 2), gemmB, 0, stream>>>(bufB,  gWt2, gb2, nullptr, gv, NPAD, 256, 128, 0);

        // ---- PairNorm * 0.25 + 0.1*vars
        zero_f32<<<1, TB, 0, stream>>>(cstats, 129);
        colsum_k<<<dim3((unsigned)cdivl(NV,128)), dim3(128), 0, stream>>>(gv, cstats, NV);
        sqsum_k <<<dim3((unsigned)cdivl(NV,128)), dim3(128), 0, stream>>>(gv, cstats, cstats+128, NV);
        pairnorm_fin<<<g1d((long)NV*FEAT), TB, 0, stream>>>(gv, cstats, cstats+128, vars, bufIn, NV);

        // ---- out MLP: 128 -> 128 -> 128 -> 1, loss
        gemm_wmma_f16<<<dim3(GX, 2), gemmB, 0, stream>>>(bufIn, oWt0, ob0, bufA, nullptr, NPAD, 128, 128, 1);
        gemm_wmma_f16<<<dim3(GX, 2), gemmB, 0, stream>>>(bufA,  oWt1, ob1, bufB, nullptr, NPAD, 128, 128, 1);
        logit_loss_k<<<dim3((unsigned)cdivl(NV,128)), dim3(128), 0, stream>>>(
            bufB, oW2, ob2, labels, out, NV, (r == ROUNDS - 1) ? 1 : 0);
    }
}
